// GumbelVectorQuantizer_56556129354020
// MI455X (gfx1250) — compile-verified, matrix operand order checked by reading, not run on hardware
//
#include <hip/hip_runtime.h>
#include <hip/hip_bf16.h>

typedef __attribute__((ext_vector_type(16))) _Float16     v16h;
typedef __attribute__((ext_vector_type(8)))  float        v8f;
typedef __attribute__((ext_vector_type(4)))  unsigned int v4u;

#define BT      32768           // 16*2048 rows
#define FSZ     768             // feature dim (K)
#define KP      (FSZ / 2)       // 384 packed f16-pairs per K-column
#define APITCH  (KP + 4)        // 388: bank spread, keeps 16B alignment
#define GV      640             // GROUPS*NUM_VARS (N)
#define NVARS   320
#define VDIM    128
#define ROWS    32              // rows per block (2 M-tiles)
#define KCH     32              // WMMA K
#define KCP     (KCH / 2)       // 16 packed pairs per WMMA K-chunk
#define NKIT    (FSZ / KCH)     // 24
#define NTW     10              // N-tiles per wave

#define SMEM_A  (ROWS * APITCH * 4)          // 49,664 B (A strip, packed f16 pairs)
#define SMEM_L  (ROWS * GV * 4)              // 81,920 B (logits, overlays A)
#define SMEM_BYTES (SMEM_L > SMEM_A ? SMEM_L : SMEM_A)

union PairPK { _Float16 h[2]; unsigned int u; };
union FragHV { v4u u[2]; v16h h; };

__device__ __forceinline__ unsigned int pk2(float a, float b) {
    PairPK p; p.h[0] = (_Float16)a; p.h[1] = (_Float16)b; return p.u;
}

// ---------------------------------------------------------------- kernel 0a
__global__ void zero_acc_kernel(float* __restrict__ acc) {
    int i = blockIdx.x * blockDim.x + threadIdx.x;
    if (i < GV) acc[i] = 0.0f;
}

// ---------------------------------------------------------------- kernel 0b
// Pre-pack W once (shared by all GEMM blocks): Wpk[c][kp] = f16pair(W[2kp][c], W[2kp+1][c]).
// This is exactly the B-fragment layout, so GEMM B loads are two global b128s.
__global__ __launch_bounds__(256)
void pack_w_kernel(const float* __restrict__ W, unsigned int* __restrict__ Wpk) {
    int idx = blockIdx.x * blockDim.x + threadIdx.x;
    if (idx >= GV * KP) return;
    int c = idx % GV, kp = idx / GV;     // consecutive t -> consecutive c (coalesced reads)
    Wpk[(size_t)c * KP + kp] =
        pk2(W[(size_t)(2 * kp) * GV + c], W[(size_t)(2 * kp + 1) * GV + c]);
}

// ---------------------------------------------------------------- kernel 1
__global__ __launch_bounds__(256)
void gemm_argmax_kernel(const float* __restrict__ x,
                        const unsigned int* __restrict__ Wpk,
                        const float* __restrict__ bias,
                        float* __restrict__ probAcc,   // [GV] global softmax-sum accumulator
                        int*   __restrict__ idxBuf)    // [BT*2] argmax indices
{
    __shared__ __align__(16) char smem[SMEM_BYTES];    // A strip, later overlaid by logits
    __shared__ float Pacc[GV];
    unsigned int* Alds = (unsigned int*)smem;          // [ROWS][APITCH]
    float*        Llds = (float*)smem;                 // [ROWS][GV] (after K loop)

    const int t    = threadIdx.x;
    const int lane = t & 31;
    const int wave = t >> 5;
    const int row0 = blockIdx.x * ROWS;

    for (int i = t; i < GV; i += 256) Pacc[i] = 0.0f;

    // stage A strip: float4 reads, f16-pair pack, one ds_store_b128 per quad
    for (int i = t; i < ROWS * (KP / 4); i += 256) {
        int r = i / (KP / 4), pq = i % (KP / 4);
        const float4* src = (const float4*)&x[(size_t)(row0 + r) * FSZ + pq * 8];
        float4 f0 = src[0], f1 = src[1];
        v4u q;
        q.x = pk2(f0.x, f0.y); q.y = pk2(f0.z, f0.w);
        q.z = pk2(f1.x, f1.y); q.w = pk2(f1.z, f1.w);
        *(v4u*)&Alds[r * APITCH + pq * 4] = q;
    }
    __syncthreads();

    v8f acc[NTW];
    for (int n = 0; n < NTW; ++n)
        for (int j = 0; j < 8; ++j) acc[n][j] = 0.0f;

    const int m_tile  = wave >> 2;      // 0..1
    const int n_grp   = wave & 3;       // 0..3
    const int half    = lane >> 4;
    const int l16     = lane & 15;
    const int pb      = half * 4;       // pair base for this lane-half
    const int arow    = (m_tile * 16 + l16) * APITCH;
    const int colbase = n_grp * NTW * 16 + l16;
    const unsigned int* wbase = Wpk + (size_t)colbase * KP + pb;

    // K loop: no barriers, B double-buffered so loads overlap WMMAs
    for (int kk = 0; kk < NKIT; ++kk) {
        const int kc = kk * KCP;

        FragHV A;
        A.u[0] = *(const v4u*)&Alds[arow + kc + pb];
        A.u[1] = *(const v4u*)&Alds[arow + kc + 8 + pb];

        FragHV Bb[2];
        Bb[0].u[0] = *(const v4u*)(wbase + kc);
        Bb[0].u[1] = *(const v4u*)(wbase + kc + 8);
#pragma unroll
        for (int nti = 0; nti < NTW; ++nti) {
            if (nti + 1 < NTW) {
                const unsigned int* p = wbase + (size_t)(nti + 1) * 16 * KP + kc;
                Bb[(nti + 1) & 1].u[0] = *(const v4u*)p;
                Bb[(nti + 1) & 1].u[1] = *(const v4u*)(p + 8);
            }
            acc[nti] = __builtin_amdgcn_wmma_f32_16x16x32_f16(
                false, A.h, false, Bb[nti & 1].h, (short)0, acc[nti], false, false);
        }
    }
    __syncthreads();   // all waves done reading Alds before logits overlay

    // store logits + bias; C/D layout: lanes0-15 -> row v, lanes16-31 -> row v+8
    for (int nti = 0; nti < NTW; ++nti) {
        const int col = (n_grp * NTW + nti) * 16 + l16;
        const float bv = bias[col];
#pragma unroll
        for (int v = 0; v < 8; ++v)
            Llds[(m_tile * 16 + v + half * 8) * GV + col] = acc[nti][v] + bv;
    }
    __syncthreads();

    // phase 2: per (row, group) wave-level argmax + softmax accumulation
    for (int task = wave; task < ROWS * 2; task += 8) {
        const int r = task >> 1;
        const int g = task & 1;
        const float* Lrow = &Llds[r * GV + g * NVARS];
        float lv[10];
        float bm = -3.4e38f;
        int   bi = 0;
#pragma unroll
        for (int j = 0; j < 10; ++j) {
            int v = lane + j * 32;
            float f = Lrow[v];
            lv[j] = f;
            if (f > bm) { bm = f; bi = v; }        // strict > keeps lowest index
        }
        for (int off = 16; off > 0; off >>= 1) {   // wave32 reduce, first-max tiebreak
            float ov = __shfl_xor(bm, off, 32);
            int   oi = __shfl_xor(bi, off, 32);
            if (ov > bm || (ov == bm && oi < bi)) { bm = ov; bi = oi; }
        }
        float s = 0.0f;
#pragma unroll
        for (int j = 0; j < 10; ++j) s += __expf(lv[j] - bm);
        for (int off = 16; off > 0; off >>= 1) s += __shfl_xor(s, off, 32);
        const float inv = 1.0f / s;
#pragma unroll
        for (int j = 0; j < 10; ++j)
            atomicAdd(&Pacc[g * NVARS + lane + j * 32], __expf(lv[j] - bm) * inv);
        if (lane == 0) idxBuf[(size_t)(row0 + r) * 2 + g] = bi;
    }
    __syncthreads();
    for (int i = t; i < GV; i += 256) atomicAdd(&probAcc[i], Pacc[i]);
}

// ---------------------------------------------------------------- kernel 2
__global__ __launch_bounds__(256)
void gather_kernel(const float* __restrict__ cb,
                   const int* __restrict__ idxBuf,
                   float* __restrict__ q)
{
    const int row = blockIdx.x;
    const int t   = threadIdx.x;
    const int g   = t >> 7;          // 0..1
    const int d   = t & 127;
    const int idx = idxBuf[row * 2 + g];
    q[(size_t)row * (2 * VDIM) + t] = cb[((size_t)g * NVARS + idx) * VDIM + d];
}

// ---------------------------------------------------------------- kernel 3
__global__ __launch_bounds__(256)
void finalize_kernel(const float* __restrict__ acc, float* __restrict__ outTail)
{
    __shared__ float sg[2];
    const int t = threadIdx.x;
    if (t < 2) sg[t] = 0.0f;
    __syncthreads();
    float local0 = 0.0f, local1 = 0.0f;
    for (int i = t; i < GV; i += 256) {
        float p = acc[i] * (1.0f / (float)BT);
        float v = p * logf(p + 1e-7f);
        if (i < NVARS) local0 += v; else local1 += v;
    }
    atomicAdd(&sg[0], local0);
    atomicAdd(&sg[1], local1);
    __syncthreads();
    if (t == 0) {
        float ppl = expf(-sg[0]) + expf(-sg[1]);
        outTail[0] = ((float)GV - ppl) / (float)GV;  // quantize_prob_ppl
        outTail[1] = 2.0f;                            // curr temp
    }
}

// ---------------------------------------------------------------- launcher
extern "C" void kernel_launch(void* const* d_in, const int* in_sizes, int n_in,
                              void* d_out, int out_size, void* d_ws, size_t ws_size,
                              hipStream_t stream) {
    const float* x  = (const float*)d_in[0];  // [16,2048,768]
    const float* W  = (const float*)d_in[1];  // [768,640]
    const float* b  = (const float*)d_in[2];  // [640]
    const float* cb = (const float*)d_in[3];  // [1,640,128]
    float* out = (float*)d_out;               // q (BT*256) ++ ppl ++ temp

    // d_ws layout: acc[640] f32 | idxBuf[BT*2] int | Wpk[GV*KP] uint (16B-aligned)
    float*        acc    = (float*)d_ws;
    int*          idxBuf = (int*)((char*)d_ws + GV * sizeof(float));
    unsigned int* Wpk    = (unsigned int*)((char*)d_ws + GV * sizeof(float)
                                           + (size_t)BT * 2 * sizeof(int));

    zero_acc_kernel<<<dim3((GV + 255) / 256), dim3(256), 0, stream>>>(acc);
    pack_w_kernel<<<dim3((GV * KP + 255) / 256), dim3(256), 0, stream>>>(W, Wpk);
    gemm_argmax_kernel<<<dim3(BT / ROWS), dim3(256), 0, stream>>>(x, Wpk, b, acc, idxBuf);
    gather_kernel<<<dim3(BT), dim3(256), 0, stream>>>(cb, idxBuf, out);
    finalize_kernel<<<dim3(1), dim3(256), 0, stream>>>(acc, out + (size_t)BT * 2 * VDIM);
}
